// FounderGNNAdvanced_31009663877559
// MI455X (gfx1250) — matrix operand order repaired and checked by reading, not compiled, not run on hardware
//
#include <hip/hip_runtime.h>
#include <hip/hip_bf16.h>

// ---------------- problem constants ----------------
constexpr int NFV = 100000;   // founders
constexpr int NUV = 5000;     // university
constexpr int NCV = 10;       // company_size
constexpr int NIV = 50;       // industry
constexpr int NRV = 20;       // role_type
constexpr int NSV = NUV + NCV + NIV + NRV; // 5080 small rows total
constexpr int HD  = 256;      // hidden
constexpr int NL  = 6;        // layers
constexpr int EV  = 250000;   // edges per relation

typedef __attribute__((ext_vector_type(16))) __bf16          v16bf;
typedef __attribute__((ext_vector_type(16))) unsigned short  v16us;
typedef __attribute__((ext_vector_type(8)))  float           v8f;

struct U4Pair { uint4 a, b; };

static __device__ __forceinline__ unsigned short f2bf(float f) {
  unsigned u = __builtin_bit_cast(unsigned, f);
  unsigned r = u + 0x7FFFu + ((u >> 16) & 1u);   // round-to-nearest-even
  return (unsigned short)(r >> 16);
}

// ---------------- fp32 -> bf16 column-major weight convert (with K padding) ----
__global__ void convert_wT(const float* __restrict__ W, unsigned short* __restrict__ WT,
                           int K, int Kp, int N) {
  int t = blockIdx.x * 256 + threadIdx.x;
  int total = N * Kp;
  if (t >= total) return;
  int n = t / Kp, k = t - n * Kp;
  float v = (k < K) ? W[(size_t)k * N + n] : 0.0f;
  WT[t] = f2bf(v);
}

// =====================================================================
// Specialized hot-path GEMM: C[M x N] += A[M x 256] * W[256 x N]
//   A fp32 row-major lda=256; W bf16 col-major [N x 256].
//   Block = 256 threads (8 waves), covers 32 rows x 128 cols.
//   A tile staged to LDS as bf16 once per block (branch-free, clamped rows),
//   each wave runs two independent WMMA accumulation chains sharing B.
//   grid = (N/128, ceil(M/32))
// =====================================================================
__global__ void __launch_bounds__(256)
gemm_acc_bf16_k256(const float* __restrict__ A, int M,
                   const unsigned short* __restrict__ WT,
                   float* __restrict__ C, int N) {
  __shared__ alignas(16) unsigned short As[32][264];  // +8 pad: 4-bank rotate per row
  int tid  = threadIdx.x;
  int wave = tid >> 5, lane = tid & 31;
  int half = lane >> 4, l16 = lane & 15;
  int tileN = blockIdx.x * 8 + wave;
  int m0 = blockIdx.y * 32;

  // ---- cooperative stage: 32 rows x 256 k of A -> LDS bf16 ----
  {
    int r   = tid >> 4;          // 0..15
    int seg = (tid & 15) << 4;   // k-start 0..240
#pragma unroll
    for (int rr = 0; rr < 2; ++rr) {
      int row  = r + rr * 16;
      int grow = m0 + row;
      const float* ap = A + (size_t)(grow < M ? grow : M - 1) * 256 + seg;
      float4 f0 = *reinterpret_cast<const float4*>(ap + 0);
      float4 f1 = *reinterpret_cast<const float4*>(ap + 4);
      float4 f2 = *reinterpret_cast<const float4*>(ap + 8);
      float4 f3 = *reinterpret_cast<const float4*>(ap + 12);
      if (grow >= M) {
        f0 = make_float4(0.f, 0.f, 0.f, 0.f); f1 = f0; f2 = f0; f3 = f0;
      }
      unsigned short* dp = &As[row][seg];
      dp[0]  = f2bf(f0.x); dp[1]  = f2bf(f0.y); dp[2]  = f2bf(f0.z); dp[3]  = f2bf(f0.w);
      dp[4]  = f2bf(f1.x); dp[5]  = f2bf(f1.y); dp[6]  = f2bf(f1.z); dp[7]  = f2bf(f1.w);
      dp[8]  = f2bf(f2.x); dp[9]  = f2bf(f2.y); dp[10] = f2bf(f2.z); dp[11] = f2bf(f2.w);
      dp[12] = f2bf(f3.x); dp[13] = f2bf(f3.y); dp[14] = f2bf(f3.z); dp[15] = f2bf(f3.w);
    }
  }
  __syncthreads();

  const unsigned short* wcol = WT + (size_t)(tileN * 16 + l16) * 256;
  v8f acc0 = {}, acc1 = {};
#pragma unroll
  for (int kb = 0; kb < 256; kb += 32) {
    int k0 = kb + half * 8;
    int k1 = kb + 16 + half * 8;
    // B fragment: two 8-bf16 runs per lane at k0 and k1 (col-major weights, L2-resident)
    U4Pair pb{*reinterpret_cast<const uint4*>(wcol + k0),
              *reinterpret_cast<const uint4*>(wcol + k1)};
    v16bf b = __builtin_bit_cast(v16bf, pb);
    __builtin_prefetch(wcol + kb + 64, 0, 1);   // global_prefetch_b8 on weight stream
    // A fragments (ISA 16-bit 16x32 layout) from LDS: two ds_load_b128 each
    U4Pair pa0{*reinterpret_cast<const uint4*>(&As[l16][k0]),
               *reinterpret_cast<const uint4*>(&As[l16][k1])};
    U4Pair pa1{*reinterpret_cast<const uint4*>(&As[16 + l16][k0]),
               *reinterpret_cast<const uint4*>(&As[16 + l16][k1])};
    v16bf a0 = __builtin_bit_cast(v16bf, pa0);
    v16bf a1 = __builtin_bit_cast(v16bf, pa1);
    acc0 = __builtin_amdgcn_wmma_f32_16x16x32_bf16(false, a0, false, b, (short)0, acc0,
                                                   false, false);
    acc1 = __builtin_amdgcn_wmma_f32_16x16x32_bf16(false, a1, false, b, (short)0, acc1,
                                                   false, false);
  }
  // ---- C read-modify-write (f32 16x16 layout) ----
  int col = tileN * 16 + l16;
#pragma unroll
  for (int r = 0; r < 8; ++r) {
    int cr0 = m0 + half * 8 + r;
    int cr1 = m0 + 16 + half * 8 + r;
    if (cr0 < M) C[(size_t)cr0 * N + col] += acc0[r];
    if (cr1 < M) C[(size_t)cr1 * N + col] += acc1[r];
  }
}

// ---------------- generic WMMA GEMM (only used for tiny proj matmuls) --------
__global__ void __launch_bounds__(256)
gemm_acc_bf16(const float* __restrict__ A, int lda, int M, int K, int Kp,
              const unsigned short* __restrict__ WT, float* __restrict__ C, int N) {
  int wave = threadIdx.x >> 5;
  int lane = threadIdx.x & 31;
  int half = lane >> 4;
  int l16  = lane & 15;
  int tileN = blockIdx.x * 8 + wave;
  int tileM = blockIdx.y;
  int row = tileM * 16 + l16;
  int col = tileN * 16 + l16;
  bool rowok = row < M;
  const float* ar = A + (size_t)(rowok ? row : 0) * lda;
  const unsigned short* wcol = WT + (size_t)col * Kp;
  v8f acc = {};
  for (int kb = 0; kb < Kp; kb += 32) {
    int k0 = kb + half * 8;
    int k1 = kb + 16 + half * 8;
    v16us au;
#pragma unroll
    for (int i = 0; i < 8; ++i) {
      float v0 = (rowok && (k0 + i) < K) ? ar[k0 + i] : 0.0f;
      float v1 = (rowok && (k1 + i) < K) ? ar[k1 + i] : 0.0f;
      au[i]     = f2bf(v0);
      au[8 + i] = f2bf(v1);
    }
    U4Pair pb{*reinterpret_cast<const uint4*>(wcol + k0),
              *reinterpret_cast<const uint4*>(wcol + k1)};
    v16bf a = __builtin_bit_cast(v16bf, au);
    v16bf b = __builtin_bit_cast(v16bf, pb);
    acc = __builtin_amdgcn_wmma_f32_16x16x32_bf16(false, a, false, b, (short)0, acc,
                                                  false, false);
  }
#pragma unroll
  for (int r = 0; r < 8; ++r) {
    int crow = tileM * 16 + half * 8 + r;
    if (crow < M) C[(size_t)crow * N + col] += acc[r];
  }
}

// ---------------- edge degree counting ----------------
__global__ void count_edges(const int* __restrict__ idx, float* __restrict__ cnt, int E) {
  int t = blockIdx.x * 256 + threadIdx.x;
  if (t < E) atomicAdd(&cnt[idx[t]], 1.0f);
}

// ---------------- segment-sum scatter: agg[dst[e]] += x[src[e]] ----------------
__global__ void scatter_add(const float* __restrict__ xs, const int* __restrict__ src,
                            const int* __restrict__ dst, float* __restrict__ agg, int E) {
  size_t t = (size_t)blockIdx.x * 256 + threadIdx.x;
  size_t e = t >> 6;                 // 64 threads per edge, 4 channels each
  if (e >= (size_t)E) return;
  int c4 = (int)(t & 63) << 2;
  int s = src[e], d = dst[e];
  float4 v = *reinterpret_cast<const float4*>(xs + (size_t)s * HD + c4);
  float* o = agg + (size_t)d * HD + c4;
  atomicAdd(o + 0, v.x); atomicAdd(o + 1, v.y);
  atomicAdd(o + 2, v.z); atomicAdd(o + 3, v.w);
}

// ---------------- agg /= max(cnt,1) ----------------
__global__ void div_count(float* __restrict__ agg, const float* __restrict__ cnt, int M) {
  size_t t = (size_t)blockIdx.x * 256 + threadIdx.x;
  if (t >= (size_t)M * HD) return;
  agg[t] /= fmaxf(cnt[t >> 8], 1.0f);   // HD == 256
}

// ---------------- sum of 4 bias vectors ----------------
__global__ void sum4(const float* a, const float* b, const float* c, const float* d,
                     float* __restrict__ o) {
  int i = threadIdx.x;
  o[i] = a[i] + b[i] + c[i] + d[i];
}

// ---------------- fused (C+bias)*scale -> GELU(exact) -> +residual ----------------
__global__ void epilogue(float* __restrict__ C, const float* __restrict__ xold,
                         const float* __restrict__ bias, float scale, int M, int res) {
  size_t t = (size_t)blockIdx.x * 256 + threadIdx.x;
  if (t >= (size_t)M * HD) return;
  int c = (int)(t & (HD - 1));
  float v = (C[t] + bias[c]) * scale;
  float g = 0.5f * v * (1.0f + erff(v * 0.70710678118654752f));
  if (res) g += xold[t];
  C[t] = g;
}

// ---------------- bias + LayerNorm (+ReLU), in place, one wave per row ----------
__global__ void __launch_bounds__(128)
ln_act(float* __restrict__ X, const float* __restrict__ bias, const float* __restrict__ g,
       const float* __restrict__ beta, int M, int D, int relu) {
  int wid = threadIdx.x >> 5, lane = threadIdx.x & 31;
  int row = blockIdx.x * 4 + wid;
  if (row >= M) return;
  int per = D >> 5;                  // 8 (D=256) or 4 (D=128)
  float vals[8];
  float* xr = X + (size_t)row * D;
  int c0 = lane * per;
  float s = 0.0f, s2 = 0.0f;
  for (int i = 0; i < per; ++i) {
    float v = xr[c0 + i] + bias[c0 + i];
    vals[i] = v; s += v; s2 += v * v;
  }
  for (int m = 16; m; m >>= 1) { s += __shfl_xor(s, m, 32); s2 += __shfl_xor(s2, m, 32); }
  float mean = s / D;
  float var  = s2 / D - mean * mean;
  float inv  = rsqrtf(var + 1e-5f);
  for (int i = 0; i < per; ++i) {
    float o = (vals[i] - mean) * inv * g[c0 + i] + beta[c0 + i];
    if (relu) o = fmaxf(o, 0.0f);
    xr[c0 + i] = o;
  }
}

// ---------------- final 128->1 projection ----------------
__global__ void __launch_bounds__(128)
dot_out(const float* __restrict__ Hm, const float* __restrict__ w,
        const float* __restrict__ b, float* __restrict__ out, int M, int D) {
  int wid = threadIdx.x >> 5, lane = threadIdx.x & 31;
  int row = blockIdx.x * 4 + wid;
  if (row >= M) return;
  const float* hr = Hm + (size_t)row * D;
  float s = 0.0f;
  for (int c = lane; c < D; c += 32) s += hr[c] * w[c];
  for (int m = 16; m; m >>= 1) s += __shfl_xor(s, m, 32);
  if (lane == 0) out[row] = s + b[0];
}

// =====================================================================
extern "C" void kernel_launch(void* const* d_in, const int* in_sizes, int n_in,
                              void* d_out, int out_size, void* d_ws, size_t ws_size,
                              hipStream_t stream) {
  (void)in_sizes; (void)out_size; (void)ws_size;
  if (n_in < 187) return;   // 13 arrays + 174 param leaves (jax pytree sorted order)

  // ---- raw inputs ----
  const float* xin[5];
  for (int i = 0; i < 5; ++i) xin[i] = (const float*)d_in[i];
  const int IN_D[5] = {64, 32, 8, 16, 8};
  const int* esrc[4] = {(const int*)d_in[5], (const int*)d_in[7],
                        (const int*)d_in[9], (const int*)d_in[11]};
  const int* edst[4] = {(const int*)d_in[6], (const int*)d_in[8],
                        (const int*)d_in[10], (const int*)d_in[12]};
  const int NDST[4]    = {NUV, NCV, NIV, NRV};         // studied_at, worked_at, in_rel, had
  const int DST_OFF[4] = {0, 5000, 5010, 5060};        // row offsets in small buffer
  auto P = [&](int leaf) -> const float* { return (const float*)d_in[leaf]; };

  // param leaf layout (sorted-key pytree flatten):
  // cls: 13..22 = b1n,b2n,g1,g2,l1.W,l1.b,l2.W,l2.b,l3.W,l3.b
  const int LBASE = 23;  // layers: 6 x 8 rel-slots x (Wl,Wr,bl)
  const int FWD_SLOT[4] = {6, 7, 1, 0};  // studied_at, worked_at, in_rel, had
  const int REV_SLOT[4] = {4, 5, 3, 2};  // rev_ counterparts
  const int PROJ_F = 171;                      // founder W,b,beta,g
  const int PROJ_S[4] = {183, 167, 175, 179};  // university, company_size, industry, role_type

  // ---- workspace carve-out ----
  char* wp = (char*)d_ws;
  auto alloc = [&](size_t bytes) -> void* {
    void* p = (void*)wp; wp += (bytes + 255) & ~(size_t)255; return p;
  };
  float* xF   = (float*)alloc((size_t)NFV * HD * 4);
  float* newF = (float*)alloc((size_t)NFV * HD * 4);
  float* aggF = (float*)alloc((size_t)NFV * HD * 4);
  float* xS   = (float*)alloc((size_t)NSV * HD * 4);
  float* newS = (float*)alloc((size_t)NSV * HD * 4);
  float* aggS = (float*)alloc((size_t)NUV * HD * 4);
  float* cntR[4]; for (int r = 0; r < 4; ++r) cntR[r] = (float*)alloc((size_t)NFV * 4);
  float* cntF[4]; for (int r = 0; r < 4; ++r) cntF[r] = (float*)alloc((size_t)NDST[r] * 4);
  float* bsum = (float*)alloc(256 * 4);
  unsigned short* WTlay = (unsigned short*)alloc((size_t)NL * 8 * 2 * 65536 * 2);
  unsigned short* WTproj[5];
  const int KPproj[5] = {64, 32, 32, 32, 32};  // founder,uni,cs,ind,rt (padded to >=32)
  WTproj[0] = (unsigned short*)alloc((size_t)256 * 64 * 2);
  for (int t = 1; t < 5; ++t) WTproj[t] = (unsigned short*)alloc((size_t)256 * 32 * 2);
  unsigned short* W1T = (unsigned short*)alloc((size_t)256 * 256 * 2);
  unsigned short* W2T = (unsigned short*)alloc((size_t)128 * 256 * 2);

  auto wt = [&](int layer, int slot, int which) -> unsigned short* {
    return WTlay + ((size_t)((layer * 8 + slot) * 2 + which)) * 65536;
  };
  auto cvt = [&](const float* W, unsigned short* WT, int K, int Kp, int N) {
    int total = N * Kp;
    convert_wT<<<(total + 255) / 256, 256, 0, stream>>>(W, WT, K, Kp, N);
  };
  auto gemm = [&](const float* A, int lda, int M, int K, int Kp,
                  const unsigned short* WT, float* C, int N) {
    if (K == 256 && lda == 256) {               // hot path: WMMA + LDS staging
      dim3 grid(N / 128, (M + 31) / 32);
      gemm_acc_bf16_k256<<<grid, 256, 0, stream>>>(A, M, WT, C, N);
    } else {                                    // tiny projection matmuls
      dim3 grid(N / 128, (M + 15) / 16);
      gemm_acc_bf16<<<grid, 256, 0, stream>>>(A, lda, M, K, Kp, WT, C, N);
    }
  };

  // ---- convert all weights to bf16 col-major (every call; no caching) ----
  for (int i = 0; i < NL; ++i)
    for (int s = 0; s < 8; ++s) {
      cvt(P(LBASE + i * 24 + s * 3 + 0), wt(i, s, 0), 256, 256, 256);  // Wl
      cvt(P(LBASE + i * 24 + s * 3 + 1), wt(i, s, 1), 256, 256, 256);  // Wr
    }
  cvt(P(PROJ_F), WTproj[0], 64, 64, 256);
  { const int KD[4] = {32, 8, 16, 8};
    for (int t = 0; t < 4; ++t) cvt(P(PROJ_S[t]), WTproj[t + 1], KD[t], 32, 256); }
  cvt(P(17), W1T, 256, 256, 256);
  cvt(P(19), W2T, 256, 256, 128);

  // ---- layer-invariant degree counts ----
  for (int r = 0; r < 4; ++r) {
    hipMemsetAsync(cntR[r], 0, (size_t)NFV * 4, stream);
    count_edges<<<(EV + 255) / 256, 256, 0, stream>>>(esrc[r], cntR[r], EV);
    hipMemsetAsync(cntF[r], 0, (size_t)NDST[r] * 4, stream);
    count_edges<<<(EV + 255) / 256, 256, 0, stream>>>(edst[r], cntF[r], EV);
  }

  // ---- input projection: relu(ln(x @ Wp + b)) ----
  hipMemsetAsync(xF, 0, (size_t)NFV * HD * 4, stream);
  gemm(xin[0], IN_D[0], NFV, IN_D[0], 64, WTproj[0], xF, 256);
  ln_act<<<(NFV + 3) / 4, 128, 0, stream>>>(xF, P(PROJ_F + 1), P(PROJ_F + 3),
                                            P(PROJ_F + 2), NFV, 256, 1);
  hipMemsetAsync(xS, 0, (size_t)NSV * HD * 4, stream);
  for (int t = 0; t < 4; ++t) {
    float* xs_t = xS + (size_t)DST_OFF[t] * HD;
    gemm(xin[t + 1], IN_D[t + 1], NDST[t], IN_D[t + 1], KPproj[t + 1], WTproj[t + 1],
         xs_t, 256);
    int pb = PROJ_S[t];
    ln_act<<<(NDST[t] + 3) / 4, 128, 0, stream>>>(xs_t, P(pb + 1), P(pb + 3), P(pb + 2),
                                                  NDST[t], 256, 1);
  }

  // ---- 6 hetero-SAGE layers ----
  for (int i = 0; i < NL; ++i) {
    int lb = LBASE + i * 24;
    sum4<<<1, 256, 0, stream>>>(P(lb + REV_SLOT[0] * 3 + 2), P(lb + REV_SLOT[1] * 3 + 2),
                                P(lb + REV_SLOT[2] * 3 + 2), P(lb + REV_SLOT[3] * 3 + 2),
                                bsum);
    hipMemsetAsync(newF, 0, (size_t)NFV * HD * 4, stream);
    hipMemsetAsync(newS, 0, (size_t)NSV * HD * 4, stream);
    for (int r = 0; r < 4; ++r) {
      int n = NDST[r];
      float* xs_r = xS + (size_t)DST_OFF[r] * HD;
      float* ns_r = newS + (size_t)DST_OFF[r] * HD;
      // forward: founder -> dst_nt
      hipMemsetAsync(aggS, 0, (size_t)n * HD * 4, stream);
      scatter_add<<<(unsigned)(((size_t)EV * 64 + 255) / 256), 256, 0, stream>>>(
          xF, esrc[r], edst[r], aggS, EV);
      div_count<<<(unsigned)(((size_t)n * HD + 255) / 256), 256, 0, stream>>>(aggS,
                                                                              cntF[r], n);
      gemm(aggS, HD, n, 256, 256, wt(i, FWD_SLOT[r], 0), ns_r, 256);
      gemm(xs_r, HD, n, 256, 256, wt(i, FWD_SLOT[r], 1), ns_r, 256);
      // reverse: dst_nt -> founder (accumulate across relations)
      hipMemsetAsync(aggF, 0, (size_t)NFV * HD * 4, stream);
      scatter_add<<<(unsigned)(((size_t)EV * 64 + 255) / 256), 256, 0, stream>>>(
          xs_r, edst[r], esrc[r], aggF, EV);
      div_count<<<(unsigned)(((size_t)NFV * HD + 255) / 256), 256, 0, stream>>>(
          aggF, cntR[r], NFV);
      gemm(aggF, HD, NFV, 256, 256, wt(i, REV_SLOT[r], 0), newF, 256);
      gemm(xF, HD, NFV, 256, 256, wt(i, REV_SLOT[r], 1), newF, 256);
    }
    // epilogues: founder (+bsum)/4, gelu, residual(i>0); dst types +bl, gelu, residual
    epilogue<<<(unsigned)(((size_t)NFV * HD + 255) / 256), 256, 0, stream>>>(
        newF, xF, bsum, 0.25f, NFV, i > 0);
    for (int r = 0; r < 4; ++r) {
      int n = NDST[r];
      float* xs_r = xS + (size_t)DST_OFF[r] * HD;
      float* ns_r = newS + (size_t)DST_OFF[r] * HD;
      epilogue<<<(unsigned)(((size_t)n * HD + 255) / 256), 256, 0, stream>>>(
          ns_r, xs_r, P(lb + FWD_SLOT[r] * 3 + 2), 1.0f, n, i > 0);
    }
    { float* t = xF; xF = newF; newF = t; }
    { float* t = xS; xS = newS; newS = t; }
  }

  // ---- classifier ----
  hipMemsetAsync(aggF, 0, (size_t)NFV * HD * 4, stream);                 // h1
  gemm(xF, HD, NFV, 256, 256, W1T, aggF, 256);
  ln_act<<<(NFV + 3) / 4, 128, 0, stream>>>(aggF, P(18), P(15), P(13), NFV, 256, 1);
  float* h2 = newF;                                                      // reuse
  hipMemsetAsync(h2, 0, (size_t)NFV * 128 * 4, stream);
  gemm(aggF, HD, NFV, 256, 256, W2T, h2, 128);
  ln_act<<<(NFV + 3) / 4, 128, 0, stream>>>(h2, P(20), P(16), P(14), NFV, 128, 1);
  dot_out<<<(NFV + 3) / 4, 128, 0, stream>>>(h2, P(21), P(22), (float*)d_out, NFV, 128);
  // emb output follows logits
  hipMemcpyAsync((float*)d_out + NFV, xF, (size_t)NFV * HD * 4,
                 hipMemcpyDeviceToDevice, stream);
}
